// Corr_10531259810235
// MI455X (gfx1250) — compile-verified, matrix operand order checked by reading, not run on hardware
//
#include <hip/hip_runtime.h>
#include <hip/hip_bf16.h>
#include <stdint.h>

// Problem constants from setup_inputs(): b=2, c=256, h=w=64
#define BATCH 2
#define CHAN  256
#define HDIM  64
#define HW    4096            // 64*64
#define TILES_1D 256          // HW/16
// sigma = h*0.05 = 3.2 ; 1/(2*sigma^2) = 1/20.48 (exact in binary)
#define INV2S2 0.048828125f

typedef __attribute__((ext_vector_type(16))) __bf16 bf16x16;
typedef __attribute__((ext_vector_type(8)))  __bf16 bf16x8;
typedef __attribute__((ext_vector_type(8)))  float  f32x8;
typedef __attribute__((ext_vector_type(4)))  float  f32x4;   // native vec for NT ld/st

union FragAB { bf16x16 v; bf16x8 h[2]; };

__device__ __forceinline__ uint16_t f2bf(float f) {
    uint32_t u = __float_as_uint(f);
    u += 0x7FFFu + ((u >> 16) & 1u);      // round-to-nearest-even
    return (uint16_t)(u >> 16);
}

// Load the 8 K-chunk fragments (K=256) for one 16-row (or 16-col) operand.
// Per ISA 16-bit 16x32 layout: lane m (0-15) holds row m, K={0..7,16..23};
// lane m+16 holds row m, K={8..15,24..31}; `base` is already lane-adjusted.
__device__ __forceinline__ void load_frags(const uint16_t* __restrict__ base,
                                           FragAB (&f)[8])
{
#pragma unroll
    for (int kk = 0; kk < 8; ++kk) {
        f[kk].h[0] = *(const bf16x8*)(base + kk * 32);
        f[kk].h[1] = *(const bf16x8*)(base + kk * 32 + 16);
    }
}

// 16x16 f32 tile = A(16x256) * B(256x16) via 8 chained bf16 WMMAs.
__device__ __forceinline__ f32x8 wmma_tile(const FragAB (&a)[8], const FragAB (&b)[8])
{
    f32x8 acc = {};
#pragma unroll
    for (int kk = 0; kk < 8; ++kk)
        acc = __builtin_amdgcn_wmma_f32_16x16x32_bf16(
                  false, a[kk].v, false, b[kk].v, (short)0, acc, false, false);
    return acc;
}

// e = exp(alpha * (1 - gauss) * aff) for C/D layout:
// element r is (row = s0 + r + 8*(lane>>4), col = t0 + (lane&15)).
__device__ __forceinline__ f32x8 mask_exp(f32x8 acc, int s0, int t0,
                                          int lane, float alpha)
{
    const int t = t0 + (lane & 15);
    const float tr = (float)(t >> 6), tc = (float)(t & 63);
    const int moff = (lane >> 4) << 3;
    f32x8 e;
#pragma unroll
    for (int r = 0; r < 8; ++r) {
        int s = s0 + r + moff;
        float dr = (float)(s >> 6) - tr;
        float dc = (float)(s & 63) - tc;
        float gauss = __expf(-(dr * dr + dc * dc) * INV2S2);
        e[r] = __expf(alpha * (1.0f - gauss) * acc[r]);
    }
    return e;
}

// ---------------------------------------------------------------------------
// Kernel 1: channel-L2 normalize, emit bf16 xn[(bb*HW+p)*CHAN + c].
// One wave per pixel, 8 waves per block.
// ---------------------------------------------------------------------------
__global__ void k_normalize(const float* __restrict__ x,
                            uint16_t* __restrict__ xn)
{
    int lane = threadIdx.x & 31;
    int wave = threadIdx.x >> 5;
    int pix  = blockIdx.x * 8 + wave;     // 0 .. BATCH*HW-1
    int bb = pix >> 12;
    int p  = pix & (HW - 1);

    float v[8];
    float s = 0.0f;
#pragma unroll
    for (int i = 0; i < 8; ++i) {
        int c = lane + (i << 5);
        float f = x[(((size_t)((bb << 8) + c)) << 12) + p];
        v[i] = f;
        s += f * f;
    }
#pragma unroll
    for (int off = 1; off < 32; off <<= 1) s += __shfl_xor(s, off, 32);

    float rn = 1.0f / fmaxf(sqrtf(s), 1e-12f);
#pragma unroll
    for (int i = 0; i < 8; ++i)
        xn[((size_t)pix << 8) + lane + (i << 5)] = f2bf(v[i] * rn);
}

// ---------------------------------------------------------------------------
// Kernel 2: GEMM pass 1 — each block owns a 16-row strip (rows s0..s0+15) of
// batch bb and all 256 t-tiles. A-fragments are stationary in 64 VGPRs; only
// B streams. Row sums R[s] accumulate in registers -> LDS -> plain store
// (no atomics; R needs no zero-init). By symmetry R also serves as col sums.
// ---------------------------------------------------------------------------
__global__ void k_rowsum(const uint16_t* __restrict__ xn,
                         const float* __restrict__ alpha_p,
                         float* __restrict__ R)
{
    __shared__ float sm_r[8][16];

    int lane = threadIdx.x & 31;
    int wave = threadIdx.x >> 5;
    int bb = blockIdx.x >> 8;
    int s0 = (blockIdx.x & 255) << 4;
    float alpha = alpha_p[0];

    const int mrow = lane & 15;
    const int ksel = (lane >> 4) << 3;
    const int moff = (lane >> 4) << 3;

    FragAB afr[8];
    load_frags(xn + (((size_t)((bb << 12) + s0 + mrow)) << 8) + ksel, afr);

    float rsum[8];
#pragma unroll
    for (int r = 0; r < 8; ++r) rsum[r] = 0.0f;

    for (int tt = wave; tt < TILES_1D; tt += 8) {   // wave-uniform loop
        int t0 = tt << 4;
        FragAB bfr[8];
        load_frags(xn + (((size_t)((bb << 12) + t0 + mrow)) << 8) + ksel, bfr);
        f32x8 e = mask_exp(wmma_tile(afr, bfr), s0, t0, lane, alpha);
#pragma unroll
        for (int r = 0; r < 8; ++r) rsum[r] += e[r];
    }

#pragma unroll
    for (int r = 0; r < 8; ++r) {
        float v = rsum[r];
        v += __shfl_xor(v, 1, 32);
        v += __shfl_xor(v, 2, 32);
        v += __shfl_xor(v, 4, 32);
        v += __shfl_xor(v, 8, 32);        // sum the 16 cols per half-wave
        if ((lane & 15) == 0)             // lane0 -> row r, lane16 -> row r+8
            sm_r[wave][r + moff] = v;
    }
    __syncthreads();

    if (threadIdx.x < 16) {
        float s = 0.0f;
#pragma unroll
        for (int w = 0; w < 8; ++w) s += sm_r[w][threadIdx.x];
        R[(bb << 12) + s0 + threadIdx.x] = s;
    }
}

__device__ __forceinline__ void push3(float& a0, float& a1, float& a2, float v) {
    if (v > a0)      { a2 = a1; a1 = a0; a0 = v; }
    else if (v > a1) { a2 = a1; a1 = v; }
    else if (v > a2) { a2 = v; }
}

// ---------------------------------------------------------------------------
// Kernel 3: GEMM pass 2 — each block owns a 4096x16 column strip of batch bb.
// B-fragments stationary in 64 VGPRs; A streams over 256 s-tiles. Writes
// g = e^2 / R[s] into x_soft (non-temporal), keeps running top-3 and S2[t],
// merges via LDS, writes val (= top3/R[t]) and Sinv[t] = 1/(S2 + 1e-8*R[t]).
// ---------------------------------------------------------------------------
__global__ void k_colpass(const uint16_t* __restrict__ xn,
                          const float* __restrict__ alpha_p,
                          const float* __restrict__ R,
                          float* __restrict__ Sinv,
                          float* __restrict__ out_val,
                          float* __restrict__ out_soft)
{
    __shared__ float sm_top[8][16][3];
    __shared__ float sm_s2[8][16];

    int lane = threadIdx.x & 31;
    int wave = threadIdx.x >> 5;
    int bb = blockIdx.x >> 8;
    int t0 = (blockIdx.x & 255) << 4;
    float alpha = alpha_p[0];

    const int mrow = lane & 15;
    const int ksel = (lane >> 4) << 3;
    const int n    = lane & 15;
    const int moff = (lane >> 4) << 3;

    FragAB bfr[8];
    load_frags(xn + (((size_t)((bb << 12) + t0 + mrow)) << 8) + ksel, bfr);

    float a0 = -1.0f, a1 = -1.0f, a2 = -1.0f;   // g >= 0 always
    float s2 = 0.0f;

    for (int st = wave; st < TILES_1D; st += 8) {   // wave-uniform loop
        int s0 = st << 4;
        FragAB afr[8];
        load_frags(xn + (((size_t)((bb << 12) + s0 + mrow)) << 8) + ksel, afr);
        f32x8 e = mask_exp(wmma_tile(afr, bfr), s0, t0, lane, alpha);
#pragma unroll
        for (int r = 0; r < 8; ++r) {
            int s = s0 + r + moff;
            float g = e[r] * e[r] / R[(bb << 12) + s];
            __builtin_nontemporal_store(
                g, &out_soft[(((size_t)((bb << 12) + s)) << 12) + t0 + n]);
            s2 += g;
            push3(a0, a1, a2, g);
        }
    }

    // merge the two lanes (L, L+16) that share column n
    float b0 = __shfl_xor(a0, 16, 32);
    float b1 = __shfl_xor(a1, 16, 32);
    float b2 = __shfl_xor(a2, 16, 32);
    s2 += __shfl_xor(s2, 16, 32);
    push3(a0, a1, a2, b0);
    push3(a0, a1, a2, b1);
    push3(a0, a1, a2, b2);

    if (lane < 16) {
        sm_top[wave][n][0] = a0;
        sm_top[wave][n][1] = a1;
        sm_top[wave][n][2] = a2;
        sm_s2[wave][n] = s2;
    }
    __syncthreads();

    if (threadIdx.x < 16) {
        int col = threadIdx.x;
        float c0 = -1.0f, c1 = -1.0f, c2 = -1.0f, st2 = 0.0f;
#pragma unroll
        for (int w = 0; w < 8; ++w) {
            st2 += sm_s2[w][col];
            push3(c0, c1, c2, sm_top[w][col][0]);
            push3(c0, c1, c2, sm_top[w][col][1]);
            push3(c0, c1, c2, sm_top[w][col][2]);
        }
        int t = t0 + col;
        float Rt = R[(bb << 12) + t];
        Sinv[(bb << 12) + t] = 1.0f / (st2 + 1e-8f * Rt);
        float rinv = 1.0f / Rt;
        // val shape (b, 3, h, w) flattened: ((bb*3 + k)*HW + t)
        out_val[(bb * 3 + 0) * HW + t] = c0 * rinv;
        out_val[(bb * 3 + 1) * HW + t] = c1 * rinv;
        out_val[(bb * 3 + 2) * HW + t] = c2 * rinv;
    }
}

// ---------------------------------------------------------------------------
// Kernel 4: streaming rescale x_soft[s,t] = g * Sinv[t]. One 16B vector per
// thread, non-temporal so the 128 MB stream doesn't evict xn/R from L2.
// ---------------------------------------------------------------------------
__global__ void k_scale(float* __restrict__ out_soft,
                        const float* __restrict__ Sinv)
{
    size_t i4 = (size_t)blockIdx.x * blockDim.x + threadIdx.x;
    f32x4* p = (f32x4*)out_soft;
    f32x4 g = __builtin_nontemporal_load(&p[i4]);
    size_t flat = i4 << 2;
    int t  = (int)(flat & (HW - 1));          // 4-aligned
    int bb = (int)(flat >> 24);               // HW*HW = 1<<24 per batch
    const f32x4 inv = *(const f32x4*)&Sinv[(bb << 12) + t];
    g *= inv;
    __builtin_nontemporal_store(g, &p[i4]);
}

// ---------------------------------------------------------------------------
extern "C" void kernel_launch(void* const* d_in, const int* in_sizes, int n_in,
                              void* d_out, int out_size, void* d_ws, size_t ws_size,
                              hipStream_t stream) {
    (void)in_sizes; (void)n_in; (void)out_size; (void)ws_size;

    const float* x       = (const float*)d_in[0];
    const float* alpha_p = (const float*)d_in[1];

    float* out      = (float*)d_out;
    float* out_val  = out;                      // BATCH*3*HW floats
    float* out_soft = out + (size_t)BATCH * 3 * HW;

    // Workspace: xn (bf16, 4 MB) | R (32 KB) | Sinv (32 KB)
    uint16_t* xn = (uint16_t*)d_ws;
    const size_t XN_ELEMS = (size_t)BATCH * HW * CHAN;
    float* R    = (float*)((char*)d_ws + XN_ELEMS * sizeof(uint16_t));
    float* Sinv = R + (size_t)BATCH * HW;

    // 1) normalize               : BATCH*HW waves, 8 per block
    k_normalize<<<(BATCH * HW) / 8, 256, 0, stream>>>(x, xn);

    // 2) row sums, row strips    : BATCH*256 strips
    k_rowsum<<<BATCH * TILES_1D, 256, 0, stream>>>(xn, alpha_p, R);

    // 3) column strips           : BATCH*256 strips
    k_colpass<<<BATCH * TILES_1D, 256, 0, stream>>>(xn, alpha_p, R, Sinv,
                                                    out_val, out_soft);

    // 4) rescale x_soft          : BATCH*HW*HW/4 vectors
    k_scale<<<(size_t)(BATCH) * HW * HW / 4 / 256, 256, 0, stream>>>(out_soft, Sinv);
}
